// TextBasedGlobalLocalUnit_13245679141081
// MI455X (gfx1250) — matrix operand order repaired and checked by reading, not compiled, not run on hardware
//
#include <hip/hip_runtime.h>

#define D_DIM 768
#define H_DIM 768
#define E_NUM 128
#define B_NUM 32
#define S_LEN 128

typedef __attribute__((ext_vector_type(16))) __bf16 v16bf;
typedef __attribute__((ext_vector_type(8)))  float  v8f;
typedef __attribute__((ext_vector_type(4)))  unsigned int uintv4;
typedef __attribute__((ext_vector_type(2)))  unsigned int uintv2;

union Frag { v16bf v; uintv4 q[2]; };
union B4   { __bf16 h[4]; uintv2 u; };

typedef __attribute__((address_space(3))) char lds_char;

// Raw LDS byte offset of a generic pointer into __shared__ (async-DMA VDST).
__device__ __forceinline__ unsigned lds_off_u32(void* p) {
  return (unsigned)(unsigned long long)(lds_char*)p;
}

// ---------------------------------------------------------------------------
// CDNA5 async tensor path: DMA one [128 rows x 64 bf16] tile (row stride 768)
// global -> LDS via GLOBAL_LOAD_ASYNC_TO_LDS_B128 (ASYNCcnt-tracked).
// 1024 x 16B transfers; 256 threads -> 4 instructions per call per wave.
// ---------------------------------------------------------------------------
__device__ __forceinline__ void async_tile_load64(__bf16* ldsTile,
                                                  const __bf16* src, int tid) {
#pragma unroll
  for (int it = 0; it < 4; ++it) {
    const int idx = tid + it * 256;
    const int s = idx >> 3, g = idx & 7;
    const unsigned dst = lds_off_u32(ldsTile + idx * 8);
    const unsigned long long ga =
        (unsigned long long)(size_t)(src + (size_t)s * 768 + (g << 3));
    asm volatile("global_load_async_to_lds_b128 %0, %1, off"
                 :: "v"(dst), "v"(ga) : "memory");
  }
}

__device__ __forceinline__ void wait_asynccnt_le8() {
  asm volatile("s_wait_asynccnt 0x8" ::: "memory");
}
__device__ __forceinline__ void wait_asynccnt_0() {
  asm volatile("s_wait_asynccnt 0x0" ::: "memory");
}

// LDS tiles: [128 rows][64 k] bf16 (row stride 128B); every fragment read is a
// 16B-aligned contiguous chunk => ds_read_b128.
// A 16x32 bf16 lane layout (ISA 7.12.2): lane<16: M=lane, K={0..7,16..23};
// lane>=16: K={8..15,24..31}.
__device__ __forceinline__ Frag load_a_frag(const __bf16* tile, int mBase, int kk,
                                            int lane) {
  const __bf16* p = tile + (size_t)(mBase + (lane & 15)) * 64 + kk;
  const int h8 = (lane >> 4) << 3;
  Frag f;
  f.q[0] = *(const uintv4*)(p + h8);
  f.q[1] = *(const uintv4*)(p + 16 + h8);
  return f;
}

// B 32x16 bf16: lane<16 -> N=lane, K={0..7,16..23}; lane>=16 -> N=lane-16, K={8..15,24..31}.
// Tile stored [N][K] so each 8 K-value group per lane is contiguous.
__device__ __forceinline__ Frag load_b_frag(const __bf16* tile, int nBase, int kk,
                                            int lane) {
  const __bf16* p = tile + (size_t)(nBase + (lane & 15)) * 64 + kk + ((lane >> 4) << 3);
  Frag f;
  f.q[0] = *(const uintv4*)(p);
  f.q[1] = *(const uintv4*)(p + 16);
  return f;
}

// 16 WMMAs on one K=64 chunk: preload all 8 B-frags per K=32 step into
// distinct register groups so the ds_load_b128s overlap the WMMA burst with
// staggered s_wait_dscnt instead of load->wait0->wmma serialization.
__device__ __forceinline__ void mma_chunk(const __bf16* curA, const __bf16* curB,
                                          v8f acc[8], int wave, int lane) {
#pragma unroll
  for (int kk = 0; kk < 64; kk += 32) {
    const Frag a = load_a_frag(curA, wave * 16, kk, lane);
    Frag bb[8];
#pragma unroll
    for (int nt = 0; nt < 8; ++nt) bb[nt] = load_b_frag(curB, nt * 16, kk, lane);
#pragma unroll
    for (int nt = 0; nt < 8; ++nt) {
      acc[nt] = __builtin_amdgcn_wmma_f32_16x16x32_bf16(
          false, a.v, false, bb[nt].v, (short)0, acc[nt], false, false);
    }
  }
}

__device__ __forceinline__ float block_reduce_sum(float v, float* red, int tid) {
  red[tid] = v;
  __syncthreads();
#pragma unroll
  for (int st = 128; st >= 1; st >>= 1) {
    if (tid < st) red[tid] += red[tid + st];
    __syncthreads();
  }
  const float r = red[0];
  __syncthreads();
  return r;
}

// ---------------------------------------------------------------------------
// Prep: f32 -> bf16 elementwise (float4 in, 4x bf16 out).
// ---------------------------------------------------------------------------
__global__ __launch_bounds__(256)
void cvt_bf16_kernel(const float* __restrict__ in, __bf16* __restrict__ out, int n4) {
  const int i = blockIdx.x * 256 + threadIdx.x;
  if (i < n4) {
    const float4 f = ((const float4*)in)[i];
    B4 t;
    t.h[0] = (__bf16)f.x; t.h[1] = (__bf16)f.y;
    t.h[2] = (__bf16)f.z; t.h[3] = (__bf16)f.w;
    ((uintv2*)out)[i] = t.u;
  }
}

// Prep: Wt[n*768+k] = (bf16)W[k*768+n], 64x64 LDS tiles.
__global__ __launch_bounds__(256)
void transpose_bf16_kernel(const float* __restrict__ W, __bf16* __restrict__ Wt) {
  __shared__ __bf16 tile[64][65];
  const int k0 = blockIdx.x * 64, n0 = blockIdx.y * 64;
  const int tid = threadIdx.x;
  for (int idx = tid; idx < 4096; idx += 256) {
    const int r = idx >> 6, c = idx & 63;
    tile[r][c] = (__bf16)W[(size_t)(k0 + r) * H_DIM + n0 + c];
  }
  __syncthreads();
  for (int idx = tid; idx < 4096; idx += 256) {
    const int r = idx >> 6, c = idx & 63;
    Wt[(size_t)(n0 + r) * D_DIM + k0 + c] = tile[c][r];
  }
}

// ---------------------------------------------------------------------------
// C[M,768] = A[M,768] @ W[768,768] + bias; A bf16 row-major, Wt bf16 [n][k].
// 256 threads / 8 waves; 128x128 tile; async double-buffered K=64 staging.
// ---------------------------------------------------------------------------
__global__ __launch_bounds__(256)
void wmma_gemm_kernel(const __bf16* __restrict__ A, const __bf16* __restrict__ Wt,
                      const float* __restrict__ bias,
                      __bf16* __restrict__ outB, float* __restrict__ outF) {
  __shared__ union SM {
    struct { __align__(16) __bf16 As[2][8192]; __align__(16) __bf16 Ws[2][8192]; } st;
    __align__(16) __bf16 Cs[128 * 128];
  } sm;                                        // 64KB

  const int m0 = blockIdx.x * 128;
  const int n0 = blockIdx.y * 128;
  const int tid = threadIdx.x;
  const int wave = tid >> 5, lane = tid & 31;
  const int hgrp = lane >> 4, nl = lane & 15;

  const __bf16* Arow = A + (size_t)m0 * D_DIM;
  const __bf16* Wrow = Wt + (size_t)n0 * D_DIM;

  async_tile_load64(sm.st.As[0], Arow, tid);
  async_tile_load64(sm.st.Ws[0], Wrow, tid);

  v8f acc[8] = {};

#pragma unroll 1
  for (int c = 0; c < 12; ++c) {
    const int cur = c & 1, nxt = cur ^ 1;
    __syncthreads();                           // buf[nxt] readers (iter c-1) done
    if (c + 1 < 12) {
      async_tile_load64(sm.st.As[nxt], Arow + (c + 1) * 64, tid);
      async_tile_load64(sm.st.Ws[nxt], Wrow + (c + 1) * 64, tid);
      wait_asynccnt_le8();                     // chunk c landed; c+1 in flight
    } else {
      wait_asynccnt_0();
    }
    __syncthreads();                           // chunk c visible to all waves
    mma_chunk(sm.st.As[cur], sm.st.Ws[cur], acc, wave, lane);
  }

  __syncthreads();                             // stage buffers dead; reuse as Cs

  if (outB) {
#pragma unroll
    for (int nt = 0; nt < 8; ++nt) {
      const int n = nt * 16 + nl;
      const float bv = bias[n0 + n];
#pragma unroll
      for (int r = 0; r < 8; ++r) {
        const int m = wave * 16 + r + 8 * hgrp;
        sm.Cs[m * 128 + n] = (__bf16)(acc[nt][r] + bv);
      }
    }
    __syncthreads();
#pragma unroll
    for (int it = 0; it < 8; ++it) {           // 2048 x 16B coalesced stores
      const int idx = tid + it * 256;
      const int m = idx >> 4, g = idx & 15;
      ((uintv4*)(outB + (size_t)(m0 + m) * H_DIM + n0))[g] = ((const uintv4*)sm.Cs)[idx];
    }
  }

  if (outF) {
#pragma unroll
    for (int nt = 0; nt < 8; ++nt) {
      const int n = n0 + nt * 16 + nl;
      const float bv = bias[n];
#pragma unroll
      for (int r = 0; r < 8; ++r) {
        const int m = m0 + wave * 16 + r + 8 * hgrp;
        outF[(size_t)m * H_DIM + n] = acc[nt][r] + bv;
      }
    }
  }
}

// ---------------------------------------------------------------------------
// Fused attention per (e,b): S = QK^T*scale (WMMA), softmax rows in registers,
// w[t] = mean_s P[s,t], ctx = w.V, LayerNorm, g2l = ctx_ln . cls_fc[e].
// ---------------------------------------------------------------------------
__global__ __launch_bounds__(256)
void attn_ctx_kernel(const __bf16* __restrict__ qB, const __bf16* __restrict__ kB,
                     const __bf16* __restrict__ vB, const float* __restrict__ clsF,
                     const float* __restrict__ ln_g, const float* __restrict__ ln_b,
                     float* __restrict__ g2l) {
  __shared__ union SM {
    struct { __align__(16) __bf16 Q[2][8192]; __align__(16) __bf16 K[2][8192]; } st;
    struct { float wt[128]; float cx[H_DIM]; float red[256]; } f;
  } sm;                                        // 64KB

  const int b = blockIdx.x, e = blockIdx.y;
  const int tid = threadIdx.x;
  const int wave = tid >> 5, lane = tid & 31;
  const int hgrp = lane >> 4, nl = lane & 15;

  const __bf16* qRow = qB + (size_t)e * (S_LEN * H_DIM);
  const __bf16* kRow = kB + (size_t)b * (S_LEN * H_DIM);

  async_tile_load64(sm.st.Q[0], qRow, tid);
  async_tile_load64(sm.st.K[0], kRow, tid);

  v8f acc[8] = {};

#pragma unroll 1
  for (int c = 0; c < 12; ++c) {
    const int cur = c & 1, nxt = cur ^ 1;
    __syncthreads();
    if (c + 1 < 12) {
      async_tile_load64(sm.st.Q[nxt], qRow + (c + 1) * 64, tid);
      async_tile_load64(sm.st.K[nxt], kRow + (c + 1) * 64, tid);
      wait_asynccnt_le8();
    } else {
      wait_asynccnt_0();
    }
    __syncthreads();
    mma_chunk(sm.st.Q[cur], sm.st.K[cur], acc, wave, lane);
  }

  __syncthreads();                             // stage buffers dead; union reuse
  if (tid < 128) sm.f.wt[tid] = 0.0f;

  // Wave holds a 16x128 score strip; row m = r + 8*hgrp spans the 16 lanes of
  // one half-wave across 8 accumulators -> xor-shuffle reductions (<16).
  const float scale = 0.036084391824351615f;   // 1/sqrt(768)
  float colp[8] = {0, 0, 0, 0, 0, 0, 0, 0};
#pragma unroll
  for (int r = 0; r < 8; ++r) {
    float mx = -3.0e38f;
#pragma unroll
    for (int nt = 0; nt < 8; ++nt) mx = fmaxf(mx, acc[nt][r]);
#pragma unroll
    for (int off = 1; off < 16; off <<= 1) mx = fmaxf(mx, __shfl_xor(mx, off, 32));
    float pv[8], sum = 0.0f;
#pragma unroll
    for (int nt = 0; nt < 8; ++nt) {
      const float t = __expf((acc[nt][r] - mx) * scale);
      pv[nt] = t;
      sum += t;
    }
#pragma unroll
    for (int off = 1; off < 16; off <<= 1) sum += __shfl_xor(sum, off, 32);
    const float inv = 1.0f / sum;
#pragma unroll
    for (int nt = 0; nt < 8; ++nt) colp[nt] += pv[nt] * inv;
  }
#pragma unroll
  for (int nt = 0; nt < 8; ++nt) colp[nt] += __shfl_xor(colp[nt], 16, 32);
  __syncthreads();                             // wt init visible everywhere
  if (hgrp == 0) {
#pragma unroll
    for (int nt = 0; nt < 8; ++nt) atomicAdd(&sm.f.wt[nt * 16 + nl], colp[nt]);
  }
  __syncthreads();
  if (tid < 128) sm.f.wt[tid] *= (1.0f / 128.0f);  // mean over query positions
  __syncthreads();

  // ctx[h] = sum_t wt[t] * V[b,t,h]; 3 h per thread, coalesced V reads.
  const __bf16* vRow = vB + (size_t)b * (S_LEN * H_DIM);
#pragma unroll
  for (int j = 0; j < 3; ++j) {
    const int h = tid + j * 256;
    float s = 0.0f;
    for (int t = 0; t < S_LEN; ++t) s += sm.f.wt[t] * (float)vRow[(size_t)t * H_DIM + h];
    sm.f.cx[h] = s;
  }
  __syncthreads();

  float part = 0.0f;
#pragma unroll
  for (int j = 0; j < 3; ++j) part += sm.f.cx[tid + j * 256];
  const float mu = block_reduce_sum(part, sm.f.red, tid) * (1.0f / (float)H_DIM);

  part = 0.0f;
#pragma unroll
  for (int j = 0; j < 3; ++j) {
    const float d = sm.f.cx[tid + j * 256] - mu;
    part += d * d;
  }
  const float var = block_reduce_sum(part, sm.f.red, tid) * (1.0f / (float)H_DIM);
  const float rstd = rsqrtf(var + 1e-5f);

  const float* cls = clsF + (size_t)e * H_DIM;
  part = 0.0f;
#pragma unroll
  for (int j = 0; j < 3; ++j) {
    const int h = tid + j * 256;
    const float y = (sm.f.cx[h] - mu) * rstd * ln_g[h] + ln_b[h];
    part += y * cls[h];
  }
  const float dotv = block_reduce_sum(part, sm.f.red, tid);
  if (tid == 0) g2l[(size_t)e * B_NUM + b] = dotv;
}

// out[b,e] = 0.5 * (g2l[e,b] + mention_cls[b] . entity_cls[e])
__global__ __launch_bounds__(128)
void finalize_kernel(const float* __restrict__ g2l, const float* __restrict__ mcls,
                     const float* __restrict__ ecls, float* __restrict__ out) {
  __shared__ float mrow[D_DIM];
  const int b = blockIdx.x;
  const int e = threadIdx.x;
  for (int d = threadIdx.x; d < D_DIM; d += 128) mrow[d] = mcls[(size_t)b * D_DIM + d];
  __syncthreads();
  const float* er = ecls + (size_t)e * D_DIM;
  float s = 0.0f;
  for (int d = 0; d < D_DIM; ++d) s += mrow[d] * er[d];
  out[(size_t)b * E_NUM + e] = 0.5f * (g2l[(size_t)e * B_NUM + b] + s);
}

extern "C" void kernel_launch(void* const* d_in, const int* in_sizes, int n_in,
                              void* d_out, int out_size, void* d_ws, size_t ws_size,
                              hipStream_t stream) {
  (void)in_sizes; (void)n_in; (void)out_size; (void)ws_size;
  const float* ecls = (const float*)d_in[0];   // [E,D]
  const float* etok = (const float*)d_in[1];   // [E,S,D]
  const float* mcls = (const float*)d_in[2];   // [B,D]
  const float* mtok = (const float*)d_in[3];   // [B,S,D]
  const float* Wq = (const float*)d_in[4];  const float* bq = (const float*)d_in[5];
  const float* Wk = (const float*)d_in[6];  const float* bk = (const float*)d_in[7];
  const float* Wv = (const float*)d_in[8];  const float* bv = (const float*)d_in[9];
  const float* Wc = (const float*)d_in[10]; const float* bc = (const float*)d_in[11];
  const float* lng = (const float*)d_in[12]; const float* lnb = (const float*)d_in[13];
  float* out = (float*)d_out;

  char* ws = (char*)d_ws;
  size_t off = 0;
  auto alloc = [&](size_t bytes) {
    char* p = ws + off;
    off += (bytes + 255) & ~(size_t)255;
    return p;
  };
  __bf16* qB    = (__bf16*)alloc((size_t)E_NUM * S_LEN * H_DIM * 2);
  __bf16* kB    = (__bf16*)alloc((size_t)B_NUM * S_LEN * H_DIM * 2);
  __bf16* vB    = (__bf16*)alloc((size_t)B_NUM * S_LEN * H_DIM * 2);
  float*  clsF  = (float*)alloc((size_t)E_NUM * H_DIM * 4);
  float*  g2l   = (float*)alloc((size_t)E_NUM * B_NUM * 4);
  __bf16* etokB = (__bf16*)alloc((size_t)E_NUM * S_LEN * D_DIM * 2);
  __bf16* mtokB = (__bf16*)alloc((size_t)B_NUM * S_LEN * D_DIM * 2);
  __bf16* eclsB = (__bf16*)alloc((size_t)E_NUM * D_DIM * 2);
  __bf16* WtQ   = (__bf16*)alloc((size_t)D_DIM * H_DIM * 2);
  __bf16* WtK   = (__bf16*)alloc((size_t)D_DIM * H_DIM * 2);
  __bf16* WtV   = (__bf16*)alloc((size_t)D_DIM * H_DIM * 2);
  __bf16* WtC   = (__bf16*)alloc((size_t)D_DIM * H_DIM * 2);

  const dim3 blk(256);

  // Prep: bf16 activations + transposed bf16 weights.
  {
    int n4;
    n4 = E_NUM * S_LEN * D_DIM / 4;
    cvt_bf16_kernel<<<dim3((n4 + 255) / 256), blk, 0, stream>>>(etok, etokB, n4);
    n4 = B_NUM * S_LEN * D_DIM / 4;
    cvt_bf16_kernel<<<dim3((n4 + 255) / 256), blk, 0, stream>>>(mtok, mtokB, n4);
    n4 = E_NUM * D_DIM / 4;
    cvt_bf16_kernel<<<dim3((n4 + 255) / 256), blk, 0, stream>>>(ecls, eclsB, n4);
    const dim3 tg(12, 12);
    transpose_bf16_kernel<<<tg, blk, 0, stream>>>(Wq, WtQ);
    transpose_bf16_kernel<<<tg, blk, 0, stream>>>(Wk, WtK);
    transpose_bf16_kernel<<<tg, blk, 0, stream>>>(Wv, WtV);
    transpose_bf16_kernel<<<tg, blk, 0, stream>>>(Wc, WtC);
  }

  // Projections (WMMA, async double-buffered staging).
  wmma_gemm_kernel<<<dim3(128, 6), blk, 0, stream>>>(etokB, WtQ, bq, qB, nullptr);
  wmma_gemm_kernel<<<dim3(32, 6),  blk, 0, stream>>>(mtokB, WtK, bk, kB, nullptr);
  wmma_gemm_kernel<<<dim3(32, 6),  blk, 0, stream>>>(mtokB, WtV, bv, vB, nullptr);
  wmma_gemm_kernel<<<dim3(1, 6),   blk, 0, stream>>>(eclsB, WtC, bc, nullptr, clsF);

  // Fused cross-attention + LayerNorm + global-local dot.
  attn_ctx_kernel<<<dim3(B_NUM, E_NUM), blk, 0, stream>>>(qB, kB, vB, clsF, lng, lnb, g2l);

  // Blend with global-global scores.
  finalize_kernel<<<dim3(B_NUM), dim3(128), 0, stream>>>(g2l, mcls, ecls, out);
}
